// YOLOv10ONNXExport_35115652612410
// MI455X (gfx1250) — matrix operand order repaired and checked by reading, not compiled, not run on hardware
//
#include <hip/hip_runtime.h>
#include <hip/hip_bf16.h>
#include <math.h>

#define NC        80
#define NCH       144        // 4*16 + 80
#define A_TOTAL   8400       // 6400 + 1600 + 400
#define MAX_DETS  300
#define CONF      0.25f
#define IMG_LIM   640.0f
#define BATCH     64

typedef __attribute__((ext_vector_type(16))) _Float16 v16h;
typedef __attribute__((ext_vector_type(8)))  float    v8f;

// -------------------------------------------------------------------------
// Decode kernel: one launch per scale. 256 threads = 8 wave32; each wave
// handles 32 anchors. DFL expected value computed with WMMA:
//   D = A(16 items x K=16 exp values, zero-padded to K=32) x B(K x 16, every
//   column = [0,1,...,15]); softmax denominator reciprocal (v_rcp_f32) is
//   computed once per item at staging time so extraction is a single multiply.
// -------------------------------------------------------------------------
__global__ __launch_bounds__(256)
void yolo_decode_kernel(const float* __restrict__ p, int HW, int W,
                        float stride, int aOff,
                        float* __restrict__ boxes,
                        float* __restrict__ best,
                        float* __restrict__ clsb)
{
    __shared__ _Float16 Esh[8 * 128 * 16];   // per-wave: 128 items x 16 exp (f16)
    __shared__ float    rdenSh[8 * 128];     // per-wave 1/denominator
    __shared__ float    distSh[8 * 128];     // per-wave DFL distances

    const int tid  = threadIdx.x;
    const int wave = tid >> 5;
    const int lane = tid & 31;
    const int b    = blockIdx.y;
    const int a0   = blockIdx.x * 256;
    const int aL   = a0 + wave * 32 + lane;          // logical anchor (may OOB)
    const int aC   = (aL < HW) ? aL : (HW - 1);      // clamped (keeps EXEC full)

    const float* pb = p + (size_t)b * NCH * HW;

    _Float16* Ew   = Esh    + wave * (128 * 16);
    float*    rden = rdenSh + wave * 128;
    float*    dst  = distSh + wave * 128;

    // ---- stage exp(logit - max) into LDS as f16; 1/sum via v_rcp_f32 ----
    #pragma unroll
    for (int s = 0; s < 4; ++s) {
        float x[16];
        #pragma unroll
        for (int k = 0; k < 16; ++k)
            x[k] = pb[(size_t)(s * 16 + k) * HW + aC];   // coalesced across lanes
        float mx = x[0];
        #pragma unroll
        for (int k = 1; k < 16; ++k) mx = fmaxf(mx, x[k]);
        const int item = lane * 4 + s;                   // anchor-local*4 + side
        float sum = 0.0f;
        #pragma unroll
        for (int k = 0; k < 16; ++k) {
            float e = __expf(x[k] - mx);
            sum += e;
            Ew[item * 16 + k] = (_Float16)e;
        }
        rden[item] = __builtin_amdgcn_rcpf(sum);         // fast v_rcp_f32
    }

    __syncthreads();

    // warm the class-logit lines while the matrix pipe works
    __builtin_prefetch(pb + (size_t)64 * HW + aC, 0, 1);

    // ---- constant B fragment: B[k][n] = k for k<16, 0 for k>=16 ----
    // 16-bit B 32x16 layout: lanes 0-15 halves 0..7 = K 0..7, lanes 16-31
    // halves 0..7 = K 8..15; halves 8..15 = K>=16 -> zero.
    v16h bfrag;
    #pragma unroll
    for (int j = 0; j < 16; ++j) bfrag[j] = (_Float16)0.0f;
    {
        #pragma unroll
        for (int j = 0; j < 8; ++j) bfrag[j] = (_Float16)(float)(j + (lane >= 16 ? 8 : 0));
    }

    const bool hi = (lane >= 16);

    // ---- 8 WMMAs: each covers 16 items (4 anchors x 4 sides) ----
    #pragma unroll
    for (int g = 0; g < 8; ++g) {
        // A 16x32 f16 layout: lane m / m+16 = row m; halves 0..7 = K 0..7 /
        // K 8..15; halves 8..15 = K>=16 -> zero.
        v16h afrag;
        #pragma unroll
        for (int j = 0; j < 16; ++j) afrag[j] = (_Float16)0.0f;
        const int itemA = g * 16 + (lane & 15);
        const int kbase = hi ? 8 : 0;
        #pragma unroll
        for (int j = 0; j < 8; ++j)
            afrag[j] = Ew[itemA * 16 + kbase + j];

        v8f acc = {0.f, 0.f, 0.f, 0.f, 0.f, 0.f, 0.f, 0.f};
        acc = __builtin_amdgcn_wmma_f32_16x16x32_f16(
                  false, afrag, false, bfrag, (short)0, acc, false, false);

        // C/D layout: VGPR r holds M=r (lanes 0-15) / M=r+8 (lanes 16-31);
        // every column identical, so lanes 0 and 16 commit rows 0-7 / 8-15.
        #pragma unroll
        for (int r = 0; r < 8; ++r) {
            const int it = g * 16 + r + (hi ? 8 : 0);
            const float d = acc[r] * rden[it];           // multiply, not divide
            if ((lane & 15) == 0) dst[it] = d;
        }
    }

    __syncthreads();

    // ---- box decode ----
    const float ltx = dst[lane * 4 + 0];
    const float lty = dst[lane * 4 + 1];
    const float rbx = dst[lane * 4 + 2];
    const float rby = dst[lane * 4 + 3];
    const float ax  = (float)(aC % W) + 0.5f;
    const float ay  = (float)(aC / W) + 0.5f;

    // ---- class max / argmax (sigmoid is monotone -> argmax on logits) ----
    float bl = -1e30f; int bc = 0;
    for (int c = 0; c < NC; ++c) {
        float v = pb[(size_t)(64 + c) * HW + aC];
        if (v > bl) { bl = v; bc = c; }
    }
    const float score = 1.0f / (1.0f + __expf(-bl));

    if (aL < HW) {
        const int ga = aOff + aL;
        float* bo = boxes + ((size_t)b * A_TOTAL + ga) * 4;
        bo[0] = (ax - ltx) * stride;
        bo[1] = (ay - lty) * stride;
        bo[2] = (ax + rbx) * stride;
        bo[3] = (ay + rby) * stride;
        best[(size_t)b * A_TOTAL + ga] = score;
        clsb[(size_t)b * A_TOTAL + ga] = (float)bc;
    }
}

// -------------------------------------------------------------------------
// Top-k kernel: one 1024-thread WG (32 wave32) per image. Masked scores live
// in LDS; 300 iterations of shfl_xor wave reduction + cross-wave LDS
// reduction; thread 0 gathers/clips/emits each detection row.
// -------------------------------------------------------------------------
__global__ __launch_bounds__(1024)
void yolo_topk_kernel(const float* __restrict__ boxes,
                      const float* __restrict__ best,
                      const float* __restrict__ clsb,
                      float* __restrict__ out)
{
    __shared__ float sc[A_TOTAL];
    __shared__ float wv[32];
    __shared__ int   wi[32];

    const int b   = blockIdx.x;
    const int tid = threadIdx.x;
    const float* bb = best + (size_t)b * A_TOTAL;

    for (int i = tid; i < A_TOTAL; i += 1024) {
        float s = bb[i];
        sc[i] = (s >= CONF) ? s : -1.0f;
    }
    __syncthreads();

    int cnt = 0;  // meaningful only on tid==0
    for (int it = 0; it < MAX_DETS; ++it) {
        float bv = -1e30f; int bi = 0x7fffffff;
        for (int i = tid; i < A_TOTAL; i += 1024) {
            float v = sc[i];
            if (v > bv) { bv = v; bi = i; }
        }
        // wave32 reduction
        #pragma unroll
        for (int off = 16; off > 0; off >>= 1) {
            float ov = __shfl_xor(bv, off, 32);
            int   oi = __shfl_xor(bi, off, 32);
            if (ov > bv || (ov == bv && oi < bi)) { bv = ov; bi = oi; }
        }
        if ((tid & 31) == 0) { wv[tid >> 5] = bv; wi[tid >> 5] = bi; }
        __syncthreads();

        if (tid < 32) {
            bv = wv[tid]; bi = wi[tid];
            #pragma unroll
            for (int off = 16; off > 0; off >>= 1) {
                float ov = __shfl_xor(bv, off, 32);
                int   oi = __shfl_xor(bi, off, 32);
                if (ov > bv || (ov == bv && oi < bi)) { bv = ov; bi = oi; }
            }
            if (tid == 0) {
                const int idx = bi;
                sc[idx] = -1e30f;   // exclude from future rounds
                const float* bx = boxes + ((size_t)b * A_TOTAL + idx) * 4;
                float x1 = fminf(fmaxf(bx[0], 0.0f), IMG_LIM);
                float y1 = fminf(fmaxf(bx[1], 0.0f), IMG_LIM);
                float x2 = fminf(fmaxf(bx[2], 0.0f), IMG_LIM);
                float y2 = fminf(fmaxf(bx[3], 0.0f), IMG_LIM);
                float s  = bb[idx];
                float so = fmaxf(s, 0.0f);
                float cv = clsb[(size_t)b * A_TOTAL + idx];
                float* o = out + ((size_t)b * MAX_DETS + it) * 6;
                o[0] = x1; o[1] = y1; o[2] = x2; o[3] = y2; o[4] = so; o[5] = cv;
                if (s >= CONF) cnt++;
            }
        }
        __syncthreads();
    }

    if (tid == 0) out[(size_t)BATCH * MAX_DETS * 6 + b] = (float)cnt;
}

// -------------------------------------------------------------------------
extern "C" void kernel_launch(void* const* d_in, const int* in_sizes, int n_in,
                              void* d_out, int out_size, void* d_ws, size_t ws_size,
                              hipStream_t stream)
{
    (void)in_sizes; (void)n_in; (void)out_size; (void)ws_size;

    const float* p8  = (const float*)d_in[0];   // (64,144,80,80)
    const float* p16 = (const float*)d_in[1];   // (64,144,40,40)
    const float* p32 = (const float*)d_in[2];   // (64,144,20,20)

    float* ws    = (float*)d_ws;
    float* boxes = ws;                                     // 64*8400*4 floats
    float* best  = boxes + (size_t)BATCH * A_TOTAL * 4;    // 64*8400
    float* clsb  = best  + (size_t)BATCH * A_TOTAL;        // 64*8400
    // total workspace: ~12.9 MB

    dim3 blk(256);
    yolo_decode_kernel<<<dim3(25, BATCH), blk, 0, stream>>>(p8, 6400, 80,  8.0f,    0, boxes, best, clsb);
    yolo_decode_kernel<<<dim3( 7, BATCH), blk, 0, stream>>>(p16, 1600, 40, 16.0f, 6400, boxes, best, clsb);
    yolo_decode_kernel<<<dim3( 2, BATCH), blk, 0, stream>>>(p32,  400, 20, 32.0f, 8000, boxes, best, clsb);

    yolo_topk_kernel<<<dim3(BATCH), dim3(1024), 0, stream>>>(boxes, best, clsb, (float*)d_out);
}
